// SimilarityTreeLSTM_49340584297236
// MI455X (gfx1250) — matrix operand order repaired and checked by reading, not compiled, not run on hardware
//
#include <hip/hip_runtime.h>
#include <hip/hip_bf16.h>

#define MEMD   512
#define NNODES 61
#define NINT   20
#define INDIM  300
#define KPAD   320
#define IOU3   1536
#define OUTD   100
#define HIDN   512
#define NCLS   5
#define VECN   20000
#define ZPAD   20032

typedef __attribute__((ext_vector_type(16))) _Float16 v16h;
typedef __attribute__((ext_vector_type(8)))  _Float16 v8h;
typedef __attribute__((ext_vector_type(8)))  float    v8f;

#define WMMA_F16(a, b, c) \
  __builtin_amdgcn_wmma_f32_16x16x32_f16(false, (a), false, (b), (short)0, (c), false, false)

// ---- fragment loaders (contiguous k order: k = 16*(lane>>4) + s), all pointers guaranteed valid ----
// A 16x32 f16 row-major: lane row m=lane&15; two contiguous 16B chunks.
__device__ __forceinline__ v16h fragA16(const _Float16* row, int kbase, int g) {
  v8h lo = *reinterpret_cast<const v8h*>(row + kbase + (g << 4));
  v8h hi = *reinterpret_cast<const v8h*>(row + kbase + 8 + (g << 4));
  return __builtin_shufflevector(lo, hi, 0, 1, 2, 3, 4, 5, 6, 7,
                                 8, 9, 10, 11, 12, 13, 14, 15);
}
// B 32x16 f16 row-major (row n=lane&15): one contiguous 32B chunk.
__device__ __forceinline__ v16h fragB16(const _Float16* row, int kbase, int g) {
  return *reinterpret_cast<const v16h*>(row + kbase + (g << 4));
}
// B fragment from an f32 row (for the two huge weights kept in f32): 4x float4 + cvt.
__device__ __forceinline__ v16h fragB32(const float* row, int kbase, int g) {
  const float4* p = reinterpret_cast<const float4*>(row + kbase + (g << 4));
  float4 q0 = p[0], q1 = p[1], q2 = p[2], q3 = p[3];
  v16h f;
  f[0]  = (_Float16)q0.x; f[1]  = (_Float16)q0.y; f[2]  = (_Float16)q0.z; f[3]  = (_Float16)q0.w;
  f[4]  = (_Float16)q1.x; f[5]  = (_Float16)q1.y; f[6]  = (_Float16)q1.z; f[7]  = (_Float16)q1.w;
  f[8]  = (_Float16)q2.x; f[9]  = (_Float16)q2.y; f[10] = (_Float16)q2.z; f[11] = (_Float16)q2.w;
  f[12] = (_Float16)q3.x; f[13] = (_Float16)q3.y; f[14] = (_Float16)q3.z; f[15] = (_Float16)q3.w;
  return f;
}

__device__ __forceinline__ float sigf(float x) { return 1.f / (1.f + expf(-x)); }

// ---- init: gather embeddings to f16 (padded), zero state + zero row ----
__global__ void k_init(const int* lin, const int* rin, const float* emb,
                       _Float16* Xh, float* H, float* C, _Float16* Hh, _Float16* Zh) {
  int tid = blockIdx.x * blockDim.x + threadIdx.x;
  int stride = gridDim.x * blockDim.x;
  for (int idx = tid; idx < 128 * KPAD; idx += stride) {
    int row = idx / KPAD, k = idx % KPAD;
    float v = 0.f;
    if (row < 2 * NNODES && k < INDIM) {
      int t = row / NNODES, n = row % NNODES;
      int w = (t ? rin : lin)[n];
      v = emb[(size_t)w * INDIM + k];
    }
    Xh[idx] = (_Float16)v;
  }
  for (int idx = tid; idx < 2 * 62 * MEMD; idx += stride) {
    H[idx] = 0.f; C[idx] = 0.f; Hh[idx] = (_Float16)0.f;
  }
  for (int idx = tid; idx < ZPAD; idx += stride) Zh[idx] = (_Float16)0.f;
}

// ---- generic f32 -> f16 weight conversion with row/K zero-padding ----
__global__ void k_cvt(const float* src, _Float16* dst, int dst_rows, int src_rows,
                      int ksrc, int kdst) {
  int tid = blockIdx.x * blockDim.x + threadIdx.x;
  int stride = gridDim.x * blockDim.x;
  int total = dst_rows * kdst;
  for (int idx = tid; idx < total; idx += stride) {
    int r = idx / kdst, k = idx % kdst;
    float v = (r < src_rows && k < ksrc) ? src[(size_t)r * ksrc + k] : 0.f;
    dst[idx] = (_Float16)v;
  }
}

// ---- batched input projections: XIOU = X@Wioux.T+bioux ; XF = X@Wfx.T+bfx ----
__global__ void k_xgemm(const _Float16* Xh, const _Float16* Wiouxh, const float* bioux,
                        const _Float16* Wfxh, const float* bfx, float* XIOU, float* XF) {
  int wave = (blockIdx.x * blockDim.x + threadIdx.x) >> 5;
  int lane = threadIdx.x & 31;
  int g = lane >> 4, mn = lane & 15;
  int mtile = wave >> 7;          // 8 M-tiles (128 rows)
  int ntile = wave & 127;         // 96 iou tiles + 32 f tiles
  bool isI = ntile < 96;
  int col = isI ? ntile * 16 + mn : (ntile - 96) * 16 + mn;
  const _Float16* Arow = Xh + (size_t)(mtile * 16 + mn) * KPAD;
  const _Float16* Brow = (isI ? Wiouxh : Wfxh) + (size_t)col * KPAD;
  v8f d = {};
  for (int kb = 0; kb < KPAD; kb += 32) {
    d = WMMA_F16(fragA16(Arow, kb, g), fragB16(Brow, kb, g), d);
  }
#pragma unroll
  for (int r = 0; r < 8; ++r) {
    int row = mtile * 16 + r + 8 * g;
    if (isI) XIOU[(size_t)row * IOU3 + col] = d[r] + bioux[col];
    else     XF[(size_t)row * MEMD + col]   = d[r] + bfx[col];
  }
}

// ---- leaves (nodes 20..60): pure elementwise LSTM cell (h_sum = 0) ----
__global__ void k_leaf(const float* XIOU, const float* biouh,
                       float* C, float* H, _Float16* Hh) {
  int tid = blockIdx.x * blockDim.x + threadIdx.x;
  int stride = gridDim.x * blockDim.x;
  const int total = 2 * 41 * MEMD;
  for (int idx = tid; idx < total; idx += stride) {
    int t = idx / (41 * MEMD);
    int rem = idx % (41 * MEMD);
    int node = 20 + rem / MEMD;
    int j = rem % MEMD;
    size_t xr = (size_t)(t * NNODES + node) * IOU3;
    float i0 = XIOU[xr + j]            + biouh[j];
    float o0 = XIOU[xr + MEMD + j]     + biouh[MEMD + j];
    float u0 = XIOU[xr + 2 * MEMD + j] + biouh[2 * MEMD + j];
    float c = sigf(i0) * tanhf(u0);
    float h = sigf(o0) * tanhf(c);
    size_t sr = (size_t)(t * 62 + node) * MEMD + j;
    C[sr] = c; H[sr] = h; Hh[sr] = (_Float16)h;
  }
}

// ---- per-node relation matmul: HSWh[t][i] = (sum_k H[child_k]) @ W_rel[rel].T ----
__global__ void k_rel(const float* H, const int* lrel, const int* rrel,
                      const float* W_rel, _Float16* HSWh, int node_start, int count) {
  __shared__ __align__(32) _Float16 hs[8][MEMD];
  __shared__ __align__(32) _Float16 zrow[MEMD];
  int wib = threadIdx.x >> 5;
  int wave = blockIdx.x * 8 + wib;
  int lane = threadIdx.x & 31;
  for (int j = threadIdx.x; j < MEMD; j += blockDim.x) zrow[j] = (_Float16)0.f;
  int total = 2 * count * 32;
  bool act = wave < total;
  int t = 0, i = 0, ntile = 0, relv = 0;
  if (act) {
    t = wave / (count * 32);
    int rem = wave % (count * 32);
    i = node_start + rem / 32;
    ntile = rem % 32;
    relv = (t ? rrel : lrel)[i];
    const float* Hb = H + (size_t)t * 62 * MEMD;
    int c0 = 3 * i + 1;
    for (int j = lane; j < MEMD; j += 32)
      hs[wib][j] = (_Float16)(Hb[(size_t)c0 * MEMD + j] + Hb[(size_t)(c0 + 1) * MEMD + j]
                            + Hb[(size_t)(c0 + 2) * MEMD + j]);
  }
  __syncthreads();
  if (!act) return;
  _Float16* out = HSWh + (size_t)(t * NINT + i) * MEMD;
  if (relv >= 45) {                       // identity relation (root)
    if (lane < 16) out[ntile * 16 + lane] = hs[wib][ntile * 16 + lane];
    return;
  }
  int g = lane >> 4, mn = lane & 15;
  const _Float16* Arow = (mn == 0) ? &hs[wib][0] : &zrow[0];   // M=1 tile
  const float* Brow = W_rel + (size_t)relv * MEMD * MEMD + (size_t)(ntile * 16 + mn) * MEMD;
  v8f d = {};
  for (int kb = 0; kb < MEMD; kb += 32) {
    d = WMMA_F16(fragA16(Arow, kb, g), fragB32(Brow, kb, g), d);
  }
  if (lane < 16) out[ntile * 16 + lane] = (_Float16)d[0];
}

// ---- batched shared-weight GEMMs per level: IOU2 = HSW@Wiouh.T ; FH = ch_h@Wfh.T ----
__global__ void k_gemm2(const _Float16* Hh, const _Float16* HSWh, const _Float16* Wiouhh,
                        const _Float16* Wfhh, const _Float16* Zh, float* IOU2, float* FH,
                        int node_start, int count) {
  int wave = (blockIdx.x * blockDim.x + threadIdx.x) >> 5;
  int lane = threadIdx.x & 31;
  int g = lane >> 4, mn = lane & 15;
  int mI = (2 * count + 15) >> 4;
  int mF = (6 * count + 15) >> 4;
  int jobsI = mI * 96;
  int total = jobsI + mF * 32;
  if (wave >= total) return;
  v8f d = {};
  if (wave < jobsI) {
    int mtile = wave / 96, ntile = wave % 96;
    int rowid = mtile * 16 + mn;
    const _Float16* Arow = Zh;
    if (rowid < 2 * count) {
      int t = rowid / count, i = node_start + rowid % count;
      Arow = HSWh + (size_t)(t * NINT + i) * MEMD;
    }
    const _Float16* Brow = Wiouhh + (size_t)(ntile * 16 + mn) * MEMD;
    for (int kb = 0; kb < MEMD; kb += 32) {
      d = WMMA_F16(fragA16(Arow, kb, g), fragB16(Brow, kb, g), d);
    }
#pragma unroll
    for (int r = 0; r < 8; ++r) {
      int rid = mtile * 16 + r + 8 * g;
      if (rid < 2 * count) {
        int tt = rid / count, ii = node_start + rid % count;
        IOU2[(size_t)(tt * NINT + ii) * IOU3 + ntile * 16 + mn] = d[r];
      }
    }
  } else {
    int w2 = wave - jobsI;
    int mtile = w2 / 32, ntile = w2 % 32;
    int rowid = mtile * 16 + mn;
    const _Float16* Arow = Zh;
    if (rowid < 6 * count) {
      int t = rowid / (3 * count);
      int rem = rowid % (3 * count);
      int i = node_start + rem / 3, ch = rem % 3;
      Arow = Hh + (size_t)(t * 62 + 3 * i + 1 + ch) * MEMD;
    }
    const _Float16* Brow = Wfhh + (size_t)(ntile * 16 + mn) * MEMD;
    for (int kb = 0; kb < MEMD; kb += 32) {
      d = WMMA_F16(fragA16(Arow, kb, g), fragB16(Brow, kb, g), d);
    }
#pragma unroll
    for (int r = 0; r < 8; ++r) {
      int rid = mtile * 16 + r + 8 * g;
      if (rid < 6 * count) {
        int t = rid / (3 * count);
        int rem = rid % (3 * count);
        int i = node_start + rem / 3, ch = rem % 3;
        FH[((size_t)(t * NINT + i) * 3 + ch) * MEMD + ntile * 16 + mn] = d[r];
      }
    }
  }
}

// ---- gate update for internal nodes at a level ----
__global__ void k_update(const float* XIOU, const float* XF, const float* IOU2,
                         const float* FH, const float* biouh, const float* bfh,
                         float* C, float* H, _Float16* Hh, int node_start, int count) {
  int tid = blockIdx.x * blockDim.x + threadIdx.x;
  int stride = gridDim.x * blockDim.x;
  int total = 2 * count * MEMD;
  for (int idx = tid; idx < total; idx += stride) {
    int t = idx / (count * MEMD);
    int rem = idx % (count * MEMD);
    int i = node_start + rem / MEMD;
    int j = rem % MEMD;
    size_t xr = (size_t)(t * NNODES + i) * IOU3;
    size_t ir = (size_t)(t * NINT + i) * IOU3;
    float i0 = XIOU[xr + j]            + IOU2[ir + j]            + biouh[j];
    float o0 = XIOU[xr + MEMD + j]     + IOU2[ir + MEMD + j]     + biouh[MEMD + j];
    float u0 = XIOU[xr + 2 * MEMD + j] + IOU2[ir + 2 * MEMD + j] + biouh[2 * MEMD + j];
    float xf = XF[(size_t)(t * NNODES + i) * MEMD + j];
    float c = sigf(i0) * tanhf(u0);
#pragma unroll
    for (int k = 0; k < 3; ++k) {
      float f = sigf(FH[((size_t)(t * NINT + i) * 3 + k) * MEMD + j] + bfh[j] + xf);
      c += f * C[(size_t)(t * 62 + 3 * i + 1 + k) * MEMD + j];
    }
    float h = sigf(o0) * tanhf(c);
    size_t sr = (size_t)(t * 62 + i) * MEMD + j;
    C[sr] = c; H[sr] = h; Hh[sr] = (_Float16)h;
  }
}

// ---- capsule projection: HJT[t][n<20][100] = H @ Wtj.T + btj ----
__global__ void k_caps(const _Float16* Hh, const _Float16* Wtjh, const float* btj,
                       const _Float16* Zh, float* HJT) {
  int wave = (blockIdx.x * blockDim.x + threadIdx.x) >> 5;
  int lane = threadIdx.x & 31;
  if (wave >= 21) return;                 // 3 M-tiles x 7 N-tiles
  int mtile = wave / 7, ntile = wave % 7;
  int g = lane >> 4, mn = lane & 15;
  int rowid = mtile * 16 + mn;
  const _Float16* Arow = Zh;
  if (rowid < 2 * NINT) {
    int t = rowid / NINT, i = rowid % NINT;
    Arow = Hh + (size_t)(t * 62 + i) * MEMD;
  }
  int col = ntile * 16 + mn;
  const _Float16* Brow = Wtjh + (size_t)col * MEMD;  // rows 100..111 pre-zeroed
  v8f d = {};
  for (int kb = 0; kb < MEMD; kb += 32) {
    d = WMMA_F16(fragA16(Arow, kb, g), fragB16(Brow, kb, g), d);
  }
#pragma unroll
  for (int r = 0; r < 8; ++r) {
    int rid = mtile * 16 + r + 8 * g;
    if (rid < 2 * NINT && col < OUTD) {
      int t = rid / NINT, i = rid % NINT;
      HJT[(size_t)(t * NINT + i) * OUTD + col] = d[r] + btj[col];
    }
  }
}

// ---- squash (uniform routing -> rank-1 v) + similarity vector (f16) ----
__global__ void k_sent(const float* HJT, _Float16* VECh) {
  __shared__ float s[2][OUTD];
  __shared__ float scale[2];
  int tid = threadIdx.x;
  for (int idx = tid; idx < 2 * OUTD; idx += blockDim.x) {
    int t = idx / OUTD, i = idx % OUTD;
    float acc = 0.f;
    for (int n = 0; n < NINT; ++n) acc += HJT[(size_t)(t * NINT + n) * OUTD + i];
    s[t][i] = acc * 0.01f;                // c = softmax(0) = 1/100
  }
  __syncthreads();
  if (tid < 2) {
    float ss = 0.f;
    for (int i = 0; i < OUTD; ++i) ss += s[tid][i] * s[tid][i];
    float mod = sqrtf(ss);
    scale[tid] = mod / (1.f + mod * mod);
  }
  __syncthreads();
  for (int idx = tid; idx < OUTD * OUTD; idx += blockDim.x) {
    int i = idx / OUTD;                   // row-major reshape: depends on row only
    float ql = scale[0] * s[0][i];
    float qr = scale[1] * s[1][i];
    VECh[idx] = (_Float16)(ql * qr);
    VECh[OUTD * OUTD + idx] = (_Float16)fabsf(ql - qr);
  }
}

// ---- classifier hidden layer: hid = sigmoid(vec @ whW.T + whb), WMMA GEMV ----
__global__ void k_hid(const _Float16* VECh, const _Float16* Zh,
                      const float* whW, const float* whb, float* HID) {
  int wave = (blockIdx.x * blockDim.x + threadIdx.x) >> 5;
  int lane = threadIdx.x & 31;
  if (wave >= 32) return;
  int g = lane >> 4, mn = lane & 15;
  const _Float16* Arow = (mn == 0) ? VECh : Zh;
  const float* Brow = whW + (size_t)(wave * 16 + mn) * VECN;
  v8f d = {};
  for (int kb = 0; kb < VECN; kb += 32) {
    d = WMMA_F16(fragA16(Arow, kb, g), fragB32(Brow, kb, g), d);
  }
  if (lane < 16) {
    int col = wave * 16 + lane;
    float z = d[0] + whb[col];
    HID[col] = 1.f / (1.f + expf(-z));
  }
}

// ---- output: logits + log_softmax, single wave ----
__global__ void k_out(const float* HID, const float* wpW, const float* wpb, float* out) {
  int lane = threadIdx.x;
  float p[NCLS] = {};
  for (int j = lane; j < HIDN; j += 32) {
    float h = HID[j];
#pragma unroll
    for (int c = 0; c < NCLS; ++c) p[c] += h * wpW[c * HIDN + j];
  }
#pragma unroll
  for (int c = 0; c < NCLS; ++c)
#pragma unroll
    for (int off = 16; off > 0; off >>= 1)
      p[c] += __shfl_down(p[c], off, 32);
  if (lane == 0) {
    float l[NCLS], m = -1e30f;
#pragma unroll
    for (int c = 0; c < NCLS; ++c) { l[c] = p[c] + wpb[c]; m = fmaxf(m, l[c]); }
    float se = 0.f;
#pragma unroll
    for (int c = 0; c < NCLS; ++c) se += expf(l[c] - m);
    float lse = m + logf(se);
#pragma unroll
    for (int c = 0; c < NCLS; ++c) out[c] = l[c] - lse;
  }
}

extern "C" void kernel_launch(void* const* d_in, const int* in_sizes, int n_in,
                              void* d_out, int out_size, void* d_ws, size_t ws_size,
                              hipStream_t stream) {
  const int*   lin   = (const int*)d_in[0];
  const int*   rin   = (const int*)d_in[1];
  const int*   lrel  = (const int*)d_in[2];
  const int*   rrel  = (const int*)d_in[3];
  const float* emb   = (const float*)d_in[4];
  const float* W_rel = (const float*)d_in[5];
  const float* Wioux = (const float*)d_in[6];
  const float* bioux = (const float*)d_in[7];
  const float* Wiouh = (const float*)d_in[8];
  const float* biouh = (const float*)d_in[9];
  const float* Wfx   = (const float*)d_in[10];
  const float* bfx   = (const float*)d_in[11];
  const float* Wfh   = (const float*)d_in[12];
  const float* bfh   = (const float*)d_in[13];
  const float* Wtj   = (const float*)d_in[14];
  const float* btj   = (const float*)d_in[15];
  const float* whW   = (const float*)d_in[16];
  const float* whb   = (const float*)d_in[17];
  const float* wpW   = (const float*)d_in[18];
  const float* wpb   = (const float*)d_in[19];

  char* base = (char*)d_ws;
  size_t off = 0;
  auto alloc = [&](size_t bytes) -> char* {
    char* p = base + off;
    off = (off + bytes + 127) & ~(size_t)127;
    return p;
  };
  // f32 scratch
  float* XIOU = (float*)alloc(128 * IOU3 * 4);
  float* XF   = (float*)alloc(128 * MEMD * 4);
  float* H    = (float*)alloc(2 * 62 * MEMD * 4);
  float* C    = (float*)alloc(2 * 62 * MEMD * 4);
  float* IOU2 = (float*)alloc(2 * NINT * IOU3 * 4);
  float* FH   = (float*)alloc(2 * NINT * 3 * MEMD * 4);
  float* HJT  = (float*)alloc(2 * NINT * OUTD * 4);
  float* HID  = (float*)alloc(HIDN * 4);
  // f16 scratch (all 128B aligned via alloc)
  _Float16* Xh      = (_Float16*)alloc(128 * KPAD * 2);
  _Float16* Wiouxh  = (_Float16*)alloc(IOU3 * KPAD * 2);
  _Float16* Wfxh    = (_Float16*)alloc(MEMD * KPAD * 2);
  _Float16* Wiouhh  = (_Float16*)alloc(IOU3 * MEMD * 2);
  _Float16* Wfhh    = (_Float16*)alloc(MEMD * MEMD * 2);
  _Float16* Wtjh    = (_Float16*)alloc(112 * MEMD * 2);
  _Float16* Hh      = (_Float16*)alloc(2 * 62 * MEMD * 2);
  _Float16* HSWh    = (_Float16*)alloc(2 * NINT * MEMD * 2);
  _Float16* VECh    = (_Float16*)alloc(VECN * 2);
  _Float16* Zh      = (_Float16*)alloc(ZPAD * 2);

  k_init<<<128, 256, 0, stream>>>(lin, rin, emb, Xh, H, C, Hh, Zh);
  k_cvt<<<1024, 256, 0, stream>>>(Wioux, Wiouxh, IOU3, IOU3, INDIM, KPAD);
  k_cvt<<<512, 256, 0, stream>>>(Wfx, Wfxh, MEMD, MEMD, INDIM, KPAD);
  k_cvt<<<1024, 256, 0, stream>>>(Wiouh, Wiouhh, IOU3, IOU3, MEMD, MEMD);
  k_cvt<<<512, 256, 0, stream>>>(Wfh, Wfhh, MEMD, MEMD, MEMD, MEMD);
  k_cvt<<<128, 256, 0, stream>>>(Wtj, Wtjh, 112, OUTD, MEMD, MEMD);

  k_xgemm<<<128, 256, 0, stream>>>(Xh, Wiouxh, bioux, Wfxh, bfx, XIOU, XF);
  k_leaf<<<82, 256, 0, stream>>>(XIOU, biouh, C, H, Hh);

  const int starts[4] = {13, 4, 1, 0};
  const int counts[4] = { 7, 9, 3, 1};
  for (int L = 0; L < 4; ++L) {
    int ns = starts[L], ct = counts[L];
    int relWaves = 2 * ct * 32;
    k_rel<<<(relWaves + 7) / 8, 256, 0, stream>>>(H, lrel, rrel, W_rel, HSWh, ns, ct);
    int mI = (2 * ct + 15) / 16, mF = (6 * ct + 15) / 16;
    int g2 = mI * 96 + mF * 32;
    k_gemm2<<<(g2 + 7) / 8, 256, 0, stream>>>(Hh, HSWh, Wiouhh, Wfhh, Zh, IOU2, FH, ns, ct);
    int upd = 2 * ct * MEMD;
    k_update<<<(upd + 255) / 256, 256, 0, stream>>>(XIOU, XF, IOU2, FH, biouh, bfh,
                                                    C, H, Hh, ns, ct);
  }

  k_caps<<<3, 256, 0, stream>>>(Hh, Wtjh, btj, Zh, HJT);
  k_sent<<<1, 256, 0, stream>>>(HJT, VECh);
  k_hid<<<4, 256, 0, stream>>>(VECh, Zh, whW, whb, HID);
  k_out<<<1, 32, 0, stream>>>(HID, wpW, wpb, (float*)d_out);
}
